// BGraphConvolution_72026601554243
// MI455X (gfx1250) — matrix operand order repaired and checked by reading, not compiled, not run on hardware
//
#include <hip/hip_runtime.h>

#define D_IN  256
#define D_OUT 128
#define ATT_H 32
#define KSTEP 32
#define XS_STRIDE 17   // uints (bf16 pairs) per row: 16 pairs + 1 pad (17 coprime with 64 banks)

typedef __attribute__((ext_vector_type(16))) __bf16 bf16x16;
typedef __attribute__((ext_vector_type(8)))  float  f32x8;

union Frag { bf16x16 v; unsigned int u[8]; };

__device__ __forceinline__ unsigned pack_bf16x2(float a, float b) {
  unsigned ua = __float_as_uint(a), ub = __float_as_uint(b);
  ua = (ua + 0x7FFFu + ((ua >> 16) & 1u)) >> 16;   // round-to-nearest-even f32 -> bf16
  ub = (ub + 0x7FFFu + ((ub >> 16) & 1u)) >> 16;
  return (ua & 0xFFFFu) | (ub << 16);
}

// ---------------------------------------------------------------------------
// Kernel 1: preA = x@Wa, preB = x@Wb via v_wmma_f32_16x16x32_bf16.
// Block = 256 threads = 8 waves; each wave owns a 16-row tile -> 128 rows/block.
// Per K-step: stage x tile (128x32) + Wa/Wb K-slices (transposed) as bf16 in LDS.
// ---------------------------------------------------------------------------
__global__ __launch_bounds__(256)
void gemm_ab_kernel(const float* __restrict__ x,
                    const float* __restrict__ Wa,
                    const float* __restrict__ Wb,
                    float* __restrict__ preA,
                    float* __restrict__ preB,
                    int n)
{
  __shared__ unsigned xs [128  * XS_STRIDE];   // x tile, [row][kpair]
  __shared__ unsigned wat[D_OUT * XS_STRIDE];  // Wa^T tile, [col][kpair]
  __shared__ unsigned wbt[D_OUT * XS_STRIDE];  // Wb^T tile, [col][kpair]

  const int tid  = threadIdx.x;
  const int wv   = tid >> 5;
  const int lane = tid & 31;
  const int ml   = lane & 15;
  const int hi   = lane >> 4;
  const int row0 = blockIdx.x * 128;

  f32x8 acca[8] = {};
  f32x8 accb[8] = {};

  for (int k0 = 0; k0 < D_IN; k0 += KSTEP) {
    // ---- stage x: thread -> (row, half of 32-k slice) as 16 floats -> 8 bf16 pairs
    {
      const int r = tid >> 1, half = tid & 1;
      const int grow = row0 + r;
      float f[16];
      if (grow < n) {
        const float4* px = (const float4*)(x + (size_t)grow * D_IN + k0 + half * 16);
        #pragma unroll
        for (int q = 0; q < 4; ++q) {
          float4 t = px[q];
          f[4*q+0] = t.x; f[4*q+1] = t.y; f[4*q+2] = t.z; f[4*q+3] = t.w;
        }
      } else {
        #pragma unroll
        for (int q = 0; q < 16; ++q) f[q] = 0.0f;
      }
      #pragma unroll
      for (int j = 0; j < 8; ++j)
        xs[r * XS_STRIDE + half * 8 + j] = pack_bf16x2(f[2*j], f[2*j+1]);
    }
    // ---- stage Wa/Wb transposed: thread -> (k-pair, 8 cols)
    {
      const int kp = tid >> 4;          // 0..15
      const int ng = (tid & 15) * 8;    // col base
      const float* wa0 = Wa + (size_t)(k0 + 2*kp    ) * D_OUT + ng;
      const float* wa1 = Wa + (size_t)(k0 + 2*kp + 1) * D_OUT + ng;
      const float* wb0 = Wb + (size_t)(k0 + 2*kp    ) * D_OUT + ng;
      const float* wb1 = Wb + (size_t)(k0 + 2*kp + 1) * D_OUT + ng;
      #pragma unroll
      for (int q = 0; q < 8; ++q) {
        wat[(ng + q) * XS_STRIDE + kp] = pack_bf16x2(wa0[q], wa1[q]);
        wbt[(ng + q) * XS_STRIDE + kp] = pack_bf16x2(wb0[q], wb1[q]);
      }
    }
    __syncthreads();

    // ---- A fragment (contiguous K, matching B): lane half selects K base 0/16
    Frag af;
    {
      const unsigned* xr = &xs[(wv * 16 + ml) * XS_STRIDE];
      const int kbh = hi * 8;           // (hi*16)/2 pair offset
      #pragma unroll
      for (int j = 0; j < 4; ++j) af.u[j]     = xr[kbh + j];
      #pragma unroll
      for (int j = 0; j < 4; ++j) af.u[4 + j] = xr[kbh + 4 + j];
    }
    // ---- B fragments + WMMA per 16-col tile
    #pragma unroll
    for (int c = 0; c < 8; ++c) {
      const int nn = c * 16 + ml;
      const int kb2h = hi * 8;          // K base 0/16 -> pair offset 0/8
      Frag bfa, bfb;
      #pragma unroll
      for (int j = 0; j < 8; ++j) {
        bfa.u[j] = wat[nn * XS_STRIDE + kb2h + j];
        bfb.u[j] = wbt[nn * XS_STRIDE + kb2h + j];
      }
      acca[c] = __builtin_amdgcn_wmma_f32_16x16x32_bf16(
          false, af.v, false, bfa.v, (short)0, acca[c], false, false);
      accb[c] = __builtin_amdgcn_wmma_f32_16x16x32_bf16(
          false, af.v, false, bfb.v, (short)0, accb[c], false, false);
    }
    __syncthreads();
  }

  // ---- store (C/D layout: VGPR r -> M = r + 8*hi, N = lane%16)
  #pragma unroll
  for (int c = 0; c < 8; ++c) {
    const int nn = c * 16 + ml;
    #pragma unroll
    for (int r = 0; r < 8; ++r) {
      const int grow = row0 + wv * 16 + hi * 8 + r;
      if (grow < n) {
        preA[(size_t)grow * D_OUT + nn] = acca[c][r];
        preB[(size_t)grow * D_OUT + nn] = accb[c][r];
      }
    }
  }
}

// ---------------------------------------------------------------------------
// Kernel 2: attention fuse. One wave per node; lane == attention head (ATT_H==32).
// P = a0*pre_a + a1*pre_al, softmax over the 2 channels.
// ---------------------------------------------------------------------------
__global__ __launch_bounds__(256)
void attn_kernel(const float* __restrict__ preA, const float* __restrict__ preB,
                 const float* __restrict__ w1, const float* __restrict__ b1,
                 const float* __restrict__ w2, float* __restrict__ P, int n)
{
  __shared__ float w1s[D_OUT * ATT_H];
  __shared__ float arow [8][D_OUT];
  __shared__ float alrow[8][D_OUT];

  const int tid = threadIdx.x, wv = tid >> 5, lane = tid & 31;
  for (int i = tid; i < D_OUT * ATT_H; i += 256) w1s[i] = w1[i];

  const int node = blockIdx.x * 8 + wv;
  const bool ok = node < n;
  float4 a4 = make_float4(0.f, 0.f, 0.f, 0.f), b4 = a4;
  if (ok) {
    a4 = *(const float4*)(preA + (size_t)node * D_OUT + lane * 4);
    b4 = *(const float4*)(preB + (size_t)node * D_OUT + lane * 4);
  }
  float4 al4;   // pre_al = pre_a + 0.5*(pre_a^2 - pre_a*pre_b)
  al4.x = a4.x + 0.5f * (a4.x * a4.x - a4.x * b4.x);
  al4.y = a4.y + 0.5f * (a4.y * a4.y - a4.y * b4.y);
  al4.z = a4.z + 0.5f * (a4.z * a4.z - a4.z * b4.z);
  al4.w = a4.w + 0.5f * (a4.w * a4.w - a4.w * b4.w);

  arow [wv][lane*4+0] = a4.x;  arow [wv][lane*4+1] = a4.y;
  arow [wv][lane*4+2] = a4.z;  arow [wv][lane*4+3] = a4.w;
  alrow[wv][lane*4+0] = al4.x; alrow[wv][lane*4+1] = al4.y;
  alrow[wv][lane*4+2] = al4.z; alrow[wv][lane*4+3] = al4.w;
  __syncthreads();

  const int t = lane;
  float h0 = b1[t], h1 = b1[t];
  for (int d = 0; d < D_OUT; ++d) {
    const float w = w1s[d * ATT_H + t];
    h0 = fmaf(arow [wv][d], w, h0);
    h1 = fmaf(alrow[wv][d], w, h1);
  }
  h0 = tanhf(h0); h1 = tanhf(h1);
  const float wt = w2[t];
  float p0 = h0 * wt, p1 = h1 * wt;
  #pragma unroll
  for (int off = 16; off > 0; off >>= 1) {
    p0 += __shfl_xor(p0, off);
    p1 += __shfl_xor(p1, off);
  }
  const float mx = fmaxf(p0, p1);
  const float e0 = __expf(p0 - mx), e1 = __expf(p1 - mx);
  const float a0 = e0 / (e0 + e1), a1 = 1.0f - a0;

  if (ok) {
    float4 o;
    o.x = a0 * a4.x + a1 * al4.x;  o.y = a0 * a4.y + a1 * al4.y;
    o.z = a0 * a4.z + a1 * al4.z;  o.w = a0 * a4.w + a1 * al4.w;
    *(float4*)(P + (size_t)node * D_OUT + lane * 4) = o;
  }
}

// ---------------------------------------------------------------------------
// SpMM kernels: one wave per edge, float4 gather, f32 atomic scatter (L2-resident).
// spmm_sq: S[row] += v*X[col]; T[row] += v*X[col]^2  (shared gather).
// ---------------------------------------------------------------------------
__global__ __launch_bounds__(256)
void spmm_sq_kernel(const int* __restrict__ rows, const int* __restrict__ cols,
                    const float* __restrict__ vals, const float* __restrict__ X,
                    float* __restrict__ S, float* __restrict__ T, int ne)
{
  const int e = blockIdx.x * 8 + (threadIdx.x >> 5);
  if (e >= ne) return;
  const int lane = threadIdx.x & 31;
  const int r = rows[e], c = cols[e];
  const float v = vals[e];
  const float4 xv = *(const float4*)(X + (size_t)c * D_OUT + lane * 4);
  float* sp = S + (size_t)r * D_OUT + lane * 4;
  float* tp = T + (size_t)r * D_OUT + lane * 4;
  unsafeAtomicAdd(sp + 0, v * xv.x);       unsafeAtomicAdd(sp + 1, v * xv.y);
  unsafeAtomicAdd(sp + 2, v * xv.z);       unsafeAtomicAdd(sp + 3, v * xv.w);
  unsafeAtomicAdd(tp + 0, v * xv.x * xv.x); unsafeAtomicAdd(tp + 1, v * xv.y * xv.y);
  unsafeAtomicAdd(tp + 2, v * xv.z * xv.z); unsafeAtomicAdd(tp + 3, v * xv.w * xv.w);
}

__global__ __launch_bounds__(256)
void spmm_scatter_kernel(const int* __restrict__ rows, const int* __restrict__ cols,
                         const float* __restrict__ vals, const float* __restrict__ X,
                         float* __restrict__ O, float scale, int ne)
{
  const int e = blockIdx.x * 8 + (threadIdx.x >> 5);
  if (e >= ne) return;
  const int lane = threadIdx.x & 31;
  const int r = rows[e], c = cols[e];
  const float v = vals[e] * scale;
  const float4 xv = *(const float4*)(X + (size_t)c * D_OUT + lane * 4);
  float* op = O + (size_t)r * D_OUT + lane * 4;
  unsafeAtomicAdd(op + 0, v * xv.x); unsafeAtomicAdd(op + 1, v * xv.y);
  unsafeAtomicAdd(op + 2, v * xv.z); unsafeAtomicAdd(op + 3, v * xv.w);
}

// ---------------------------------------------------------------------------
// Elementwise helpers
// ---------------------------------------------------------------------------
__global__ __launch_bounds__(256)
void zero_kernel(float4* __restrict__ p, int n4)
{
  const int i = blockIdx.x * 256 + threadIdx.x;
  if (i < n4) p[i] = make_float4(0.f, 0.f, 0.f, 0.f);
}

// D = (accum ? D : 0) + sign * 0.5 * (S*S - T)
__global__ __launch_bounds__(256)
void combine_kernel(const float4* __restrict__ S, const float4* __restrict__ T,
                    float4* __restrict__ D, int accum, float sign, int n4)
{
  const int i = blockIdx.x * 256 + threadIdx.x;
  if (i >= n4) return;
  const float4 s = S[i], t = T[i];
  float4 d;
  d.x = sign * 0.5f * (s.x * s.x - t.x);
  d.y = sign * 0.5f * (s.y * s.y - t.y);
  d.z = sign * 0.5f * (s.z * s.z - t.z);
  d.w = sign * 0.5f * (s.w * s.w - t.w);
  if (accum) {
    const float4 dd = D[i];
    d.x += dd.x; d.y += dd.y; d.z += dd.z; d.w += dd.w;
  }
  D[i] = d;
}

__global__ __launch_bounds__(256)
void relu_kernel(float4* __restrict__ p, int n4)
{
  const int i = blockIdx.x * 256 + threadIdx.x;
  if (i >= n4) return;
  float4 v = p[i];
  v.x = fmaxf(v.x, 0.f); v.y = fmaxf(v.y, 0.f);
  v.z = fmaxf(v.z, 0.f); v.w = fmaxf(v.w, 0.f);
  p[i] = v;
}

// ---------------------------------------------------------------------------
// Launch pipeline
// ---------------------------------------------------------------------------
extern "C" void kernel_launch(void* const* d_in, const int* in_sizes, int n_in,
                              void* d_out, int out_size, void* d_ws, size_t ws_size,
                              hipStream_t stream)
{
  const float* x    = (const float*)d_in[0];
  const float* Wa   = (const float*)d_in[1];
  const float* Wb   = (const float*)d_in[2];
  // d_in[3] = Wc: computed-but-unused in the reference; skipped.
  const float* w1   = (const float*)d_in[4];
  const float* b1   = (const float*)d_in[5];
  const float* w2   = (const float*)d_in[6];
  const int*   rows = (const int*)d_in[7];
  const int*   cols = (const int*)d_in[8];
  const float* vals = (const float*)d_in[9];
  float* out = (float*)d_out;

  const int n  = in_sizes[0] / D_IN;      // nodes
  const int ne = in_sizes[7] / 7;         // edges per support
  const size_t NP = (size_t)n * D_OUT;

  float* P  = (float*)d_ws;               // pre_sup
  float* A  = P + NP;                     // preA, later S
  float* B  = A + NP;                     // preB, later T
  float* Dd = B + NP;                     // bilinear diff
  float* S = A;
  float* T = B;

  const int n4    = (int)(NP / 4);
  const int gElem = (n4 + 255) / 256;
  const int gEdge = (ne + 7) / 8;
  const int gGemm = (n + 127) / 128;
  const int gAttn = (n + 7) / 8;
  const dim3 blk(256);

  // Dense path
  gemm_ab_kernel<<<gGemm, blk, 0, stream>>>(x, Wa, Wb, A, B, n);
  attn_kernel<<<gAttn, blk, 0, stream>>>(A, B, w1, b1, w2, P, n);

  // out = 0.5 * spmm(0, P)  (ALPHA = 0.5)
  zero_kernel<<<gElem, blk, 0, stream>>>((float4*)out, n4);
  spmm_scatter_kernel<<<gEdge, blk, 0, stream>>>(rows, cols, vals, P, out, 0.5f, ne);

  // d1 = bp(1) - bp(3); out += 0.25 * spmm(5, d1)
  zero_kernel<<<gElem, blk, 0, stream>>>((float4*)S, n4);
  zero_kernel<<<gElem, blk, 0, stream>>>((float4*)T, n4);
  spmm_sq_kernel<<<gEdge, blk, 0, stream>>>(rows + 1*(size_t)ne, cols + 1*(size_t)ne,
                                            vals + 1*(size_t)ne, P, S, T, ne);
  combine_kernel<<<gElem, blk, 0, stream>>>((const float4*)S, (const float4*)T,
                                            (float4*)Dd, 0, 1.0f, n4);
  zero_kernel<<<gElem, blk, 0, stream>>>((float4*)S, n4);
  zero_kernel<<<gElem, blk, 0, stream>>>((float4*)T, n4);
  spmm_sq_kernel<<<gEdge, blk, 0, stream>>>(rows + 3*(size_t)ne, cols + 3*(size_t)ne,
                                            vals + 3*(size_t)ne, P, S, T, ne);
  combine_kernel<<<gElem, blk, 0, stream>>>((const float4*)S, (const float4*)T,
                                            (float4*)Dd, 1, -1.0f, n4);
  spmm_scatter_kernel<<<gEdge, blk, 0, stream>>>(rows + 5*(size_t)ne, cols + 5*(size_t)ne,
                                                 vals + 5*(size_t)ne, Dd, out, 0.25f, ne);

  // d2 = bp(2) - bp(4); out += 0.25 * spmm(6, d2)
  zero_kernel<<<gElem, blk, 0, stream>>>((float4*)S, n4);
  zero_kernel<<<gElem, blk, 0, stream>>>((float4*)T, n4);
  spmm_sq_kernel<<<gEdge, blk, 0, stream>>>(rows + 2*(size_t)ne, cols + 2*(size_t)ne,
                                            vals + 2*(size_t)ne, P, S, T, ne);
  combine_kernel<<<gElem, blk, 0, stream>>>((const float4*)S, (const float4*)T,
                                            (float4*)Dd, 0, 1.0f, n4);
  zero_kernel<<<gElem, blk, 0, stream>>>((float4*)S, n4);
  zero_kernel<<<gElem, blk, 0, stream>>>((float4*)T, n4);
  spmm_sq_kernel<<<gEdge, blk, 0, stream>>>(rows + 4*(size_t)ne, cols + 4*(size_t)ne,
                                            vals + 4*(size_t)ne, P, S, T, ne);
  combine_kernel<<<gElem, blk, 0, stream>>>((const float4*)S, (const float4*)T,
                                            (float4*)Dd, 1, -1.0f, n4);
  spmm_scatter_kernel<<<gEdge, blk, 0, stream>>>(rows + 6*(size_t)ne, cols + 6*(size_t)ne,
                                                 vals + 6*(size_t)ne, Dd, out, 0.25f, ne);

  relu_kernel<<<gElem, blk, 0, stream>>>((float4*)out, n4);
}
